// Speller_16793322127469
// MI455X (gfx1250) — matrix operand order repaired and checked by reading, not compiled, not run on hardware
//
#include <hip/hip_runtime.h>

// ---------------------------------------------------------------------------
// CDNA5 / gfx1250 Speller (attention LSTM decoder) — bf16 WMMA implementation
// ---------------------------------------------------------------------------

typedef __bf16 bf16;
typedef bf16  bf16x2  __attribute__((ext_vector_type(2)));
typedef bf16  bf16x8  __attribute__((ext_vector_type(8)));
typedef bf16  bf16x16 __attribute__((ext_vector_type(16)));
typedef float f32x8   __attribute__((ext_vector_type(8)));

#define UNITS 512
#define BATCH 32
#define TSTEP 128
#define SLEN  512
#define VOCAB 46
#define N4U   2048   // 4 * UNITS

// ---------------------------------------------------------------------------
// Pack a [K,N] fp32 weight matrix into WMMA-native bf16 layout:
// per 32x16 (KxN) tile, 32 lanes x 16 bf16 contiguous, matching
// the A loads' contiguous-k order, not AMD's table layout (lane<16: K 0..15,
// lane>=16: K 16..31; element j -> K = half*16+j, N = nt*16 + (lane&15)).
// ---------------------------------------------------------------------------
__global__ void pack_b_kernel(const float* __restrict__ src, bf16* __restrict__ dst,
                              int K, int N)
{
    size_t p = (size_t)blockIdx.x * blockDim.x + threadIdx.x;
    if (p >= (size_t)K * N) return;
    int    j    = (int)(p & 15);
    int    lane = (int)((p >> 4) & 31);
    size_t tile = p >> 9;
    int    KT   = K >> 5;
    int    kt   = (int)(tile % KT);
    int    nt   = (int)(tile / KT);
    int    half = lane >> 4;
    int    l    = lane & 15;
    int    n    = nt * 16 + l;
    int    k    = kt * 32 + half * 16 + j;
    dst[p] = (bf16)src[(size_t)k * N + n];
}

// y [B,T,D] fp32  ->  ybf [(t*B+b), D] bf16 (time-major rows for per-step GEMM)
__global__ void conv_y_kernel(const float* __restrict__ y, bf16* __restrict__ ybf)
{
    size_t i = (size_t)blockIdx.x * blockDim.x + threadIdx.x;
    if (i >= (size_t)TSTEP * BATCH * UNITS) return;
    int d   = (int)(i & 511);
    int row = (int)(i >> 9);
    int t   = row >> 5;    // row / BATCH
    int b   = row & 31;    // row % BATCH
    ybf[i] = (bf16)y[((size_t)b * TSTEP + t) * UNITS + d];
}

// x fp32 -> bf16, flat (halves per-step attention L2 traffic, 16MB resident)
__global__ void conv_x_kernel(const float* __restrict__ x, bf16* __restrict__ xbf)
{
    size_t i = (size_t)blockIdx.x * blockDim.x + threadIdx.x;
    if (i < (size_t)BATCH * SLEN * UNITS) xbf[i] = (bf16)x[i];
}

__global__ void zero_state_kernel(float* __restrict__ c, bf16* __restrict__ hbf)
{
    int i = blockIdx.x * blockDim.x + threadIdx.x;
    if (i < BATCH * UNITS) { c[i] = 0.0f; hbf[i] = (bf16)0.0f; }
}

// ---------------------------------------------------------------------------
// Per-step GEMM:  Z[M,N] = A0[M,K0] @ B0 + A1[M,K1] @ B1 + bias
// A row-major bf16; B packed by pack_b_kernel; Z fp32.
// One wave per 16x16 C tile; 4 waves per block.
// ---------------------------------------------------------------------------
__device__ __forceinline__ f32x8 wmma_accum_pass(
    f32x8 acc, const bf16* __restrict__ A, const bf16* __restrict__ Bp,
    int K, int m, int nt, int half, int lane)
{
    const int   KT    = K >> 5;
    const bf16* arow  = A + (size_t)m * K + half * 16;
    const bf16* bbase = Bp + ((size_t)nt * KT * 32 + lane) * 16;
    for (int kt = 0; kt < KT; ++kt) {
        const bf16* ap = arow + kt * 32;
        const bf16* bp = bbase + (size_t)kt * 512;      // 32 lanes * 16 elems
        __builtin_prefetch(bp + 512, 0, 1);             // next packed B tile
        bf16x8 alo = *(const bf16x8*)(ap);
        bf16x8 ahi = *(const bf16x8*)(ap + 8);
        bf16x8 blo = *(const bf16x8*)(bp);
        bf16x8 bhi = *(const bf16x8*)(bp + 8);
        bf16x16 a, b;
        #pragma unroll
        for (int i = 0; i < 8; ++i) {
            a[i] = alo[i]; a[i + 8] = ahi[i];
            b[i] = blo[i]; b[i + 8] = bhi[i];
        }
        acc = __builtin_amdgcn_wmma_f32_16x16x32_bf16(
                  false, a, false, b, (short)0, acc, false, false);
    }
    return acc;
}

__global__ __launch_bounds__(128) void gemm_step_kernel(
    const bf16* __restrict__ A0, const bf16* __restrict__ B0p,
    const bf16* __restrict__ A1, const bf16* __restrict__ B1p,
    const float* __restrict__ bias, float* __restrict__ Z,
    int M, int N, int K0, int K1)
{
    const int lane = threadIdx.x & 31;
    const int wave = threadIdx.x >> 5;
    const int Mt = M >> 4, Nt = N >> 4;
    int tile = blockIdx.x * 4 + wave;
    if (tile >= Mt * Nt) return;
    const int mt   = tile % Mt;
    const int nt   = tile / Mt;
    const int half = lane >> 4;
    const int l    = lane & 15;
    const int m    = mt * 16 + l;

    f32x8 acc = {0.f, 0.f, 0.f, 0.f, 0.f, 0.f, 0.f, 0.f};
    acc = wmma_accum_pass(acc, A0, B0p, K0, m, nt, half, lane);   // input proj
    acc = wmma_accum_pass(acc, A1, B1p, K1, m, nt, half, lane);   // recurrent proj

    const int n  = nt * 16 + l;
    const float bv = bias ? bias[n] : 0.0f;
    #pragma unroll
    for (int r = 0; r < 8; ++r) {
        int mm = mt * 16 + half * 8 + r;                // C/D layout: M = half*8 + r
        Z[(size_t)mm * N + n] = acc[r] + bv;
    }
}

// ---------------------------------------------------------------------------
// LSTM gate nonlinearity + state update (Keras order i,f,g,o).
// ---------------------------------------------------------------------------
__device__ __forceinline__ float sigf(float x) { return 1.0f / (1.0f + expf(-x)); }

__global__ void lstm_gates_kernel(const float* __restrict__ z, float* __restrict__ c,
                                  float* __restrict__ hf32, bf16* __restrict__ hbf,
                                  bf16* __restrict__ seqrow)
{
    int i = blockIdx.x * blockDim.x + threadIdx.x;
    if (i >= BATCH * UNITS) return;
    int b = i >> 9, u = i & 511;
    const float* zb = z + (size_t)b * N4U;
    float ig = sigf(zb[u]);
    float fg = sigf(zb[512 + u]);
    float g  = tanhf(zb[1024 + u]);
    float o  = sigf(zb[1536 + u]);
    float cn = fg * c[i] + ig * g;
    float hn = o * tanhf(cn);
    c[i] = cn;
    if (hf32)   hf32[i]   = hn;
    hbf[i] = (bf16)hn;
    if (seqrow) seqrow[i] = (bf16)hn;
}

// ---------------------------------------------------------------------------
// Fused attention step: gates + scores + softmax + context + residual.
// One block per batch row (32 blocks x 256 threads = 8 waves).
//   - gates keep c_new in registers, h_s in LDS
//   - scores: wave-cooperative dot over bf16 x rows (coalesced b128 loads,
//     wave32 shfl_xor reduction)
//   - softmax + weighted context sum over bf16 x, fp32 accumulate
//   - h <- h_s + ctx (bf16 for next recurrent GEMM), c <- c_new + ctx,
//     ctx row stored bf16 as LSTM1 input.
// ---------------------------------------------------------------------------
__global__ __launch_bounds__(256) void attn_fused_kernel(
    const float* __restrict__ z, const bf16* __restrict__ xbf,
    float* __restrict__ c, bf16* __restrict__ hbf,
    bf16* __restrict__ ctxbf, int t)
{
    __shared__ float hsm[UNITS];
    __shared__ float p[SLEN];
    __shared__ float red[256];
    __shared__ float m_s, inv_s;

    const int b   = blockIdx.x;
    const int tid = threadIdx.x;
    const int u0  = tid * 2, u1 = u0 + 1;
    const float* zb = z + (size_t)b * N4U;

    // ---- gates for units u0,u1 (c_new held in registers) ----
    float cn0, cn1;
    {
        float ig = sigf(zb[u0]);
        float fg = sigf(zb[512 + u0]);
        float g  = tanhf(zb[1024 + u0]);
        float o  = sigf(zb[1536 + u0]);
        cn0 = fg * c[b * UNITS + u0] + ig * g;
        hsm[u0] = o * tanhf(cn0);

        ig = sigf(zb[u1]);
        fg = sigf(zb[512 + u1]);
        g  = tanhf(zb[1024 + u1]);
        o  = sigf(zb[1536 + u1]);
        cn1 = fg * c[b * UNITS + u1] + ig * g;
        hsm[u1] = o * tanhf(cn1);
    }
    __syncthreads();

    // ---- scores[s] = dot(h_s, x[b,s,:]) : wave-cooperative ----
    const bf16* xb  = xbf + (size_t)b * SLEN * UNITS;
    const int  wave = tid >> 5, lane = tid & 31;
    for (int s = wave; s < SLEN; s += 8) {
        const bf16* xr = xb + (size_t)s * UNITS + lane * 16;
        bf16x8 v0 = *(const bf16x8*)(xr);
        bf16x8 v1 = *(const bf16x8*)(xr + 8);
        float acc = 0.f;
        #pragma unroll
        for (int i = 0; i < 8; ++i) {
            acc += (float)v0[i] * hsm[lane * 16 + i];
            acc += (float)v1[i] * hsm[lane * 16 + 8 + i];
        }
        #pragma unroll
        for (int off = 16; off > 0; off >>= 1) acc += __shfl_xor(acc, off, 32);
        if (lane == 0) p[s] = acc;            // raw score
    }
    __syncthreads();

    // ---- softmax over S=512 ----
    float a0 = p[tid], a1 = p[tid + 256];
    red[tid] = fmaxf(a0, a1);
    __syncthreads();
    for (int o = 128; o > 0; o >>= 1) {
        if (tid < o) red[tid] = fmaxf(red[tid], red[tid + o]);
        __syncthreads();
    }
    if (tid == 0) m_s = red[0];
    __syncthreads();
    float e0 = expf(a0 - m_s), e1 = expf(a1 - m_s);
    p[tid] = e0; p[tid + 256] = e1;
    red[tid] = e0 + e1;
    __syncthreads();
    for (int o = 128; o > 0; o >>= 1) {
        if (tid < o) red[tid] += red[tid + o];
        __syncthreads();
    }
    if (tid == 0) inv_s = 1.0f / red[0];
    __syncthreads();

    // ---- context: ctx[u] = sum_s p[s] * x[b,s,u] (coalesced bf16x2) ----
    const float inv = inv_s;
    float acc0 = 0.f, acc1 = 0.f;
    for (int s = 0; s < SLEN; ++s) {
        bf16x2 v = *(const bf16x2*)(xb + (size_t)s * UNITS + u0);
        float ps = p[s];
        acc0 += ps * (float)v[0];
        acc1 += ps * (float)v[1];
    }
    float ctx0 = acc0 * inv, ctx1 = acc1 * inv;

    const int idx0 = b * UNITS + u0, idx1 = b * UNITS + u1;
    hbf[idx0] = (bf16)(hsm[u0] + ctx0);       // residual into h (next GEMM A)
    hbf[idx1] = (bf16)(hsm[u1] + ctx1);
    c[idx0] = cn0 + ctx0;                     // residual into c
    c[idx1] = cn1 + ctx1;
    bf16* crow = ctxbf + ((size_t)t * BATCH + b) * UNITS;
    crow[u0] = (bf16)ctx0;                    // scan output (LSTM1 input)
    crow[u1] = (bf16)ctx1;
}

// logits = h @ Wd + bd ; softmax over V=46. One block per batch row.
__global__ void dense_softmax_kernel(const float* __restrict__ h,
                                     const float* __restrict__ Wd,
                                     const float* __restrict__ bd,
                                     float* __restrict__ out)
{
    __shared__ float lg[64];
    __shared__ float m_s, s_s;
    int b = blockIdx.x, v = threadIdx.x;
    float acc = -1e30f;
    if (v < VOCAB) {
        acc = bd[v];
        const float* hr = h + (size_t)b * UNITS;
        for (int u = 0; u < UNITS; ++u) acc += hr[u] * Wd[(size_t)u * VOCAB + v];
    }
    lg[v] = acc;
    __syncthreads();
    if (v == 0) {
        float mm = -1e30f;
        for (int j = 0; j < VOCAB; ++j) mm = fmaxf(mm, lg[j]);
        m_s = mm;
    }
    __syncthreads();
    float e = (v < VOCAB) ? expf(lg[v] - m_s) : 0.0f;
    lg[v] = e;
    __syncthreads();
    if (v == 0) {
        float ss = 0.f;
        for (int j = 0; j < VOCAB; ++j) ss += lg[j];
        s_s = ss;
    }
    __syncthreads();
    if (v < VOCAB) out[b * VOCAB + v] = e / s_s;
}

// ---------------------------------------------------------------------------
// Host orchestration
// ---------------------------------------------------------------------------
extern "C" void kernel_launch(void* const* d_in, const int* in_sizes, int n_in,
                              void* d_out, int out_size, void* d_ws, size_t ws_size,
                              hipStream_t stream)
{
    (void)in_sizes; (void)n_in; (void)out_size; (void)ws_size;
    const float* x   = (const float*)d_in[0];   // [B,S,U]
    const float* y   = (const float*)d_in[1];   // [B,T,D]
    const float* W_a = (const float*)d_in[2];
    const float* R_a = (const float*)d_in[3];
    const float* b_a = (const float*)d_in[4];
    const float* W1  = (const float*)d_in[5];
    const float* R1  = (const float*)d_in[6];
    const float* b1  = (const float*)d_in[7];
    const float* W2  = (const float*)d_in[8];
    const float* R2  = (const float*)d_in[9];
    const float* b2  = (const float*)d_in[10];
    const float* Wd  = (const float*)d_in[11];
    const float* bd  = (const float*)d_in[12];
    float* out = (float*)d_out;

    // carve workspace (deterministic, no allocation)
    char* w = (char*)d_ws;
    auto carve = [&](size_t bytes) -> void* {
        void* p = (void*)w;
        w += (bytes + 255) & ~(size_t)255;
        return p;
    };
    const size_t WB = (size_t)UNITS * N4U * sizeof(bf16);              // 2 MB
    bf16* Wa_p  = (bf16*)carve(WB);
    bf16* Ra_p  = (bf16*)carve(WB);
    bf16* W1_p  = (bf16*)carve(WB);
    bf16* R1_p  = (bf16*)carve(WB);
    bf16* W2_p  = (bf16*)carve(WB);
    bf16* R2_p  = (bf16*)carve(WB);
    const size_t SEQB = (size_t)TSTEP * BATCH * UNITS * sizeof(bf16);  // 4 MB
    bf16* ybf   = (bf16*)carve(SEQB);
    bf16* ctxbf = (bf16*)carve(SEQB);
    bf16* hs1bf = (bf16*)carve(SEQB);
    bf16* xbf   = (bf16*)carve((size_t)BATCH * SLEN * UNITS * sizeof(bf16)); // 16 MB
    const size_t STF = (size_t)BATCH * UNITS * sizeof(float);
    const size_t STB = (size_t)BATCH * UNITS * sizeof(bf16);
    float* cA   = (float*)carve(STF);
    bf16*  hAbf = (bf16*)carve(STB);
    float* c1   = (float*)carve(STF);
    bf16*  h1bf = (bf16*)carve(STB);
    float* c2   = (float*)carve(STF);
    float* h2f  = (float*)carve(STF);
    bf16*  h2bf = (bf16*)carve(STB);
    float* z    = (float*)carve((size_t)BATCH * N4U * sizeof(float));

    // 1) pack weights into WMMA-native layout (parallel, once per call)
    {
        const int n = UNITS * N4U, blk = 256, grd = n / blk;
        pack_b_kernel<<<grd, blk, 0, stream>>>(W_a, Wa_p, UNITS, N4U);
        pack_b_kernel<<<grd, blk, 0, stream>>>(R_a, Ra_p, UNITS, N4U);
        pack_b_kernel<<<grd, blk, 0, stream>>>(W1,  W1_p, UNITS, N4U);
        pack_b_kernel<<<grd, blk, 0, stream>>>(R1,  R1_p, UNITS, N4U);
        pack_b_kernel<<<grd, blk, 0, stream>>>(W2,  W2_p, UNITS, N4U);
        pack_b_kernel<<<grd, blk, 0, stream>>>(R2,  R2_p, UNITS, N4U);
    }
    // 2) activations -> bf16
    conv_y_kernel<<<(TSTEP * BATCH * UNITS) / 256, 256, 0, stream>>>(y, ybf);
    conv_x_kernel<<<(BATCH * SLEN * UNITS) / 256, 256, 0, stream>>>(x, xbf);

    const int GEMM_GRID = (2 * (N4U / 16)) / 4;   // 256 tiles / 4 waves = 64 blocks
    const int EW_GRID   = (BATCH * UNITS) / 256;  // 64 blocks

    // 3) Attention LSTM scan: 2 launches per step (GEMM + fused attention)
    zero_state_kernel<<<EW_GRID, 256, 0, stream>>>(cA, hAbf);
    for (int t = 0; t < TSTEP; ++t) {
        gemm_step_kernel<<<GEMM_GRID, 128, 0, stream>>>(
            ybf + (size_t)t * BATCH * UNITS, Wa_p, hAbf, Ra_p, b_a, z,
            BATCH, N4U, UNITS, UNITS);
        attn_fused_kernel<<<BATCH, 256, 0, stream>>>(z, xbf, cA, hAbf, ctxbf, t);
    }

    // 4) LSTM layer 1 scan (input = ctx sequence)
    zero_state_kernel<<<EW_GRID, 256, 0, stream>>>(c1, h1bf);
    for (int t = 0; t < TSTEP; ++t) {
        gemm_step_kernel<<<GEMM_GRID, 128, 0, stream>>>(
            ctxbf + (size_t)t * BATCH * UNITS, W1_p, h1bf, R1_p, b1, z,
            BATCH, N4U, UNITS, UNITS);
        lstm_gates_kernel<<<EW_GRID, 256, 0, stream>>>(
            z, c1, nullptr, h1bf, hs1bf + (size_t)t * BATCH * UNITS);
    }

    // 5) LSTM layer 2 scan (input = layer-1 hidden sequence); keep last h fp32
    zero_state_kernel<<<EW_GRID, 256, 0, stream>>>(c2, h2bf);
    for (int t = 0; t < TSTEP; ++t) {
        gemm_step_kernel<<<GEMM_GRID, 128, 0, stream>>>(
            hs1bf + (size_t)t * BATCH * UNITS, W2_p, h2bf, R2_p, b2, z,
            BATCH, N4U, UNITS, UNITS);
        lstm_gates_kernel<<<EW_GRID, 256, 0, stream>>>(z, c2, h2f, h2bf, nullptr);
    }

    // 6) dense + softmax
    dense_softmax_kernel<<<BATCH, 64, 0, stream>>>(h2f, Wd, bd, out);
}
